// NetVLAD_80376017977860
// MI455X (gfx1250) — matrix operand order repaired and checked by reading, not compiled, not run on hardware
//
#include <hip/hip_runtime.h>
#include <hip/hip_bf16.h>
#include <math.h>

typedef _Float16 h16;
typedef __attribute__((ext_vector_type(16))) _Float16 v16h;
typedef __attribute__((ext_vector_type(8)))  _Float16 v8h;
typedef __attribute__((ext_vector_type(8)))  float    v8f;

#define NB 16
#define TT 8192
#define CC 128
#define KK 64
#define EPS_BN   1e-5f
#define EPS_NRM  1e-12f
#define NEG_FILL (-1e18f)

// workspace layout (bytes)
#define OFF_XNH   0ull                                   // f16 xn, [N][C][T]   32 MB
#define OFF_AH    (OFF_XNH  + (size_t)NB*CC*TT*2)        // f16 a,  [N][K][T]   16 MB
#define OFF_VL    (OFF_AH   + (size_t)NB*KK*TT*2)        // f32 vlad partials  512 KB
#define OFF_ASUM  (OFF_VL   + (size_t)NB*KK*CC*4)        // f32 sum_t a          4 KB
#define OFF_W2F   (OFF_ASUM + (size_t)NB*KK*4)           // f16 w2 fragments    16 KB
#define OFF_CONST (OFF_W2F  + (size_t)16*512*2)          // f32 folded consts

// const-region float offsets
#define CST_W1T  0     // [128*3] depthwise taps (middle column of 3x3)
#define CST_S1   384   // [128] bn1 scale
#define CST_SH1  512   // [128] bn1 shift
#define CST_S2   640   // [64]  bn2 scale
#define CST_SH2  704   // [64]  bn2 shift
#define CST_B2   768   // [64]  conv2 bias

static __device__ inline v8f wmma_f16(v16h a, v16h b, v8f c) {
  return __builtin_amdgcn_wmma_f32_16x16x32_f16(false, a, false, b, (short)0, c, false, false);
}

// ---------------- init: fold BN, extract taps, pre-pack w2 into B-fragment order
__global__ void k_init(const float* __restrict__ conv1_w,
                       const float* __restrict__ g1, const float* __restrict__ b1,
                       const float* __restrict__ m1, const float* __restrict__ v1,
                       const float* __restrict__ conv2_w, const float* __restrict__ cb2,
                       const float* __restrict__ g2, const float* __restrict__ b2_,
                       const float* __restrict__ m2, const float* __restrict__ v2,
                       h16* __restrict__ w2f, float* __restrict__ cst) {
  int tid = threadIdx.x;
  // 16 fragments (cc in 0..3 x kt in 0..3), 512 halves each, per-lane 16-half runs.
  // B element (K=c, N=k) = conv2_w[k][c]; lane<16: N=lane, K=0..7,16..23 of chunk; lane>=16: K=8..15,24..31.
  for (int idx = tid; idx < 16 * 512; idx += 256) {
    int frag = idx >> 9, within = idx & 511;
    int lane = within >> 4, j = within & 15;
    int ccb = frag >> 2, kt = frag & 3;
    int k = kt * 16 + (lane & 15);
    int c = ccb * 32 + ((lane >> 4) << 3) + (j & 7) + ((j >> 3) << 4);
    w2f[idx] = (h16)conv2_w[k * CC + c];
  }
  if (tid < CC) {
    float s = g1[tid] * rsqrtf(v1[tid] + EPS_BN);
    cst[CST_S1 + tid]  = s;
    cst[CST_SH1 + tid] = b1[tid] - m1[tid] * s;
    // conv1_w shape [C][1][3][3]; W=1 with pad 1 -> only middle column (kw=1) contributes
    cst[CST_W1T + tid * 3 + 0] = conv1_w[tid * 9 + 0 * 3 + 1];
    cst[CST_W1T + tid * 3 + 1] = conv1_w[tid * 9 + 1 * 3 + 1];
    cst[CST_W1T + tid * 3 + 2] = conv1_w[tid * 9 + 2 * 3 + 1];
  }
  if (tid < KK) {
    float s = g2[tid] * rsqrtf(v2[tid] + EPS_BN);
    cst[CST_S2 + tid]  = s;
    cst[CST_SH2 + tid] = b2_[tid] - m2[tid] * s;
    cst[CST_B2 + tid]  = cb2[tid];
  }
}

__global__ void k_zero(float* __restrict__ p, int n) {
  int i = blockIdx.x * blockDim.x + threadIdx.x;
  if (i < n) p[i] = 0.0f;
}

// ---------------- per-descriptor L2 normalize; write f16 xn in [N][C][T]
__global__ void __launch_bounds__(CC) k_norm(const float* __restrict__ x, h16* __restrict__ xnh) {
  int bid = blockIdx.x;
  int n = bid / TT, t = bid % TT;
  int c = threadIdx.x;
  __shared__ float red[CC];
  float v = x[((size_t)(n * TT + t)) * CC + c];
  red[c] = v * v;
  __syncthreads();
  for (int s = 64; s > 0; s >>= 1) { if (c < s) red[c] += red[c + s]; __syncthreads(); }
  float inv = 1.0f / fmaxf(sqrtf(red[0]), EPS_NRM);
  xnh[((size_t)(n * CC + c)) * TT + t] = (h16)(v * inv);
}

// ---------------- depthwise conv + BN1 + ReLU + [TxC]@[CxK] WMMA + BN2 + ReLU + mask + softmax
__global__ void __launch_bounds__(32) k_assign(const h16* __restrict__ xnh,
                                               const h16* __restrict__ w2f,
                                               const float* __restrict__ cst,
                                               const int* __restrict__ length,
                                               h16* __restrict__ ah,
                                               float* __restrict__ asum) {
  int n = blockIdx.y;
  int t0 = blockIdx.x * 16;
  int lane = threadIdx.x;
  int len = length[n];

  __shared__ h16   ldx[18 * CC];   // rows = t0-1 .. t0+16
  __shared__ h16   ldh[16 * CC];   // h1 tile [t][c]
  __shared__ float ldsa[16 * KK];  // soft-assign tile [t][k]

  // stage xn tile (with halo) from c-major global layout
  for (int idx = lane; idx < 18 * CC; idx += 32) {
    int tl = idx >> 7, c = idx & 127;
    int tg = t0 + tl - 1;
    float v = 0.0f;
    if (tg >= 0 && tg < TT) v = (float)xnh[((size_t)(n * CC + c)) * TT + tg];
    ldx[idx] = (h16)v;
  }
  __syncthreads();

  // depthwise 3-tap conv + folded BN1 + ReLU
  for (int idx = lane; idx < 16 * CC; idx += 32) {
    int tl = idx >> 7, c = idx & 127;
    float h = cst[CST_W1T + c * 3 + 0] * (float)ldx[tl * CC + c]
            + cst[CST_W1T + c * 3 + 1] * (float)ldx[(tl + 1) * CC + c]
            + cst[CST_W1T + c * 3 + 2] * (float)ldx[(tl + 2) * CC + c];
    h = fmaxf(cst[CST_S1 + c] * h + cst[CST_SH1 + c], 0.0f);
    ldh[idx] = (h16)h;
  }
  __syncthreads();

  // D[t,k] = h1[t,c] @ w2t[c,k] : 4 c-chunks x 4 k-tiles of WMMA
  v8f acc[4];
  #pragma unroll
  for (int kt = 0; kt < 4; ++kt) acc[kt] = v8f{};
  int tr = lane & 15;
  #pragma unroll
  for (int ccb = 0; ccb < 4; ++ccb) {
    int cbase = ccb * 32 + ((lane >> 4) << 3);
    v8h a0 = *(const v8h*)&ldh[tr * CC + cbase];
    v8h a1 = *(const v8h*)&ldh[tr * CC + cbase + 16];
    v16h a;
    #pragma unroll
    for (int i = 0; i < 8; ++i) { a[i] = a0[i]; a[8 + i] = a1[i]; }
    #pragma unroll
    for (int kt = 0; kt < 4; ++kt) {
      v16h b = *(const v16h*)&w2f[(ccb * 4 + kt) * 512 + lane * 16];
      acc[kt] = wmma_f16(a, b, acc[kt]);
    }
  }

  // epilogue: bias + BN2 + ReLU + length-mask -> LDS [t][k]
  #pragma unroll
  for (int kt = 0; kt < 4; ++kt) {
    int k = kt * 16 + (lane & 15);
    #pragma unroll
    for (int r = 0; r < 8; ++r) {
      int tl = r + ((lane >> 4) << 3);
      float sa = fmaxf(cst[CST_S2 + k] * (acc[kt][r] + cst[CST_B2 + k]) + cst[CST_SH2 + k], 0.0f);
      if (t0 + tl >= len) sa = NEG_FILL;
      ldsa[tl * KK + k] = sa;
    }
  }
  __syncthreads();

  // softmax over K=64 per t-row (lanes 0..15 own rows)
  if (lane < 16) {
    int row = lane;
    float m = NEG_FILL;
    for (int k = 0; k < KK; ++k) m = fmaxf(m, ldsa[row * KK + k]);
    float s = 0.0f;
    for (int k = 0; k < KK; ++k) {
      float e = __expf(ldsa[row * KK + k] - m);
      ldsa[row * KK + k] = e;
      s += e;
    }
    float inv = 1.0f / s;
    for (int k = 0; k < KK; ++k) ldsa[row * KK + k] *= inv;
  }
  __syncthreads();

  // write a as f16 in [N][K][T] (A-fragment friendly), plus sum_t a per cluster
  #pragma unroll
  for (int kk = 0; kk < 2; ++kk) {
    int k = lane + kk * 32;
    float s = 0.0f;
    v16h hv;
    #pragma unroll
    for (int r = 0; r < 16; ++r) {
      float v = ldsa[r * KK + k];
      s += v;
      hv[r] = (h16)v;
    }
    *(v16h*)&ah[((size_t)(n * KK + k)) * TT + t0] = hv;
    atomicAdd(&asum[n * KK + k], s);
  }
}

// ---------------- vlad partial GEMM: D[k,c] += a[k,t] * xn[c,t] over a 512-t slab
__global__ void __launch_bounds__(128) k_vlad(const h16* __restrict__ ah,
                                              const h16* __restrict__ xnh,
                                              float* __restrict__ vl) {
  int n = blockIdx.y;
  int t0 = blockIdx.x * 512;
  int wave = threadIdx.x >> 5;
  int lane = threadIdx.x & 31;
  int k0 = wave * 16;

  v8f acc[8];
  #pragma unroll
  for (int cn = 0; cn < 8; ++cn) acc[cn] = v8f{};

  for (int tc = 0; tc < 16; ++tc) {
    int t = t0 + tc * 32;
    // A fragment: rows = clusters, K-dim = t
    int kA = k0 + (lane & 15);
    int koff = (lane >> 4) << 4;
    const h16* pA = ah + ((size_t)(n * KK + kA)) * TT + t + koff;
    v16h a = *(const v16h*)pA;
    #pragma unroll
    for (int cn = 0; cn < 8; ++cn) {
      int c = cn * 16 + (lane & 15);
      const h16* pB = xnh + ((size_t)(n * CC + c)) * TT + t + ((lane >> 4) << 4);
      v16h b = *(const v16h*)pB;
      acc[cn] = wmma_f16(a, b, acc[cn]);
    }
  }

  #pragma unroll
  for (int cn = 0; cn < 8; ++cn) {
    #pragma unroll
    for (int r = 0; r < 8; ++r) {
      int k = k0 + r + ((lane >> 4) << 3);
      int c = cn * 16 + (lane & 15);
      atomicAdd(&vl[((size_t)(n * KK + k)) * CC + c], acc[cn][r]);
    }
  }
}

// ---------------- centroid subtraction + intra L2 + global L2
__global__ void __launch_bounds__(CC) k_final(const float* __restrict__ vl,
                                              const float* __restrict__ asum,
                                              const float* __restrict__ centroids,
                                              float* __restrict__ out) {
  int n = blockIdx.x;
  int c = threadIdx.x;
  __shared__ float red[CC];
  float gss = 0.0f;
  for (int k = 0; k < KK; ++k) {
    float v = vl[((size_t)(n * KK + k)) * CC + c] - asum[n * KK + k] * centroids[k * CC + c];
    red[c] = v * v;
    __syncthreads();
    for (int s = 64; s > 0; s >>= 1) { if (c < s) red[c] += red[c + s]; __syncthreads(); }
    float inv = 1.0f / fmaxf(sqrtf(red[0]), EPS_NRM);
    __syncthreads();
    float r = v * inv;
    out[(size_t)n * KK * CC + k * CC + c] = r;
    gss += r * r;
  }
  red[c] = gss;
  __syncthreads();
  for (int s = 64; s > 0; s >>= 1) { if (c < s) red[c] += red[c + s]; __syncthreads(); }
  float ginv = 1.0f / fmaxf(sqrtf(red[0]), EPS_NRM);
  __syncthreads();
  for (int k = 0; k < KK; ++k) {
    size_t o = (size_t)n * KK * CC + k * CC + c;
    out[o] = out[o] * ginv;
  }
}

extern "C" void kernel_launch(void* const* d_in, const int* in_sizes, int n_in,
                              void* d_out, int out_size, void* d_ws, size_t ws_size,
                              hipStream_t stream) {
  (void)in_sizes; (void)n_in; (void)out_size; (void)ws_size;
  const float* x        = (const float*)d_in[0];
  const float* conv1_w  = (const float*)d_in[1];
  const float* bn1_g    = (const float*)d_in[2];
  const float* bn1_b    = (const float*)d_in[3];
  const float* bn1_m    = (const float*)d_in[4];
  const float* bn1_v    = (const float*)d_in[5];
  const float* conv2_w  = (const float*)d_in[6];
  const float* conv2_b  = (const float*)d_in[7];
  const float* bn2_g    = (const float*)d_in[8];
  const float* bn2_b    = (const float*)d_in[9];
  const float* bn2_m    = (const float*)d_in[10];
  const float* bn2_v    = (const float*)d_in[11];
  const float* centroids= (const float*)d_in[12];
  const int*   length   = (const int*)d_in[13];

  char* ws = (char*)d_ws;
  h16*   xnh  = (h16*)(ws + OFF_XNH);
  h16*   ah   = (h16*)(ws + OFF_AH);
  float* vl   = (float*)(ws + OFF_VL);
  float* asum = (float*)(ws + OFF_ASUM);
  h16*   w2f  = (h16*)(ws + OFF_W2F);
  float* cst  = (float*)(ws + OFF_CONST);
  float* out  = (float*)d_out;

  k_init<<<1, 256, 0, stream>>>(conv1_w, bn1_g, bn1_b, bn1_m, bn1_v,
                                conv2_w, conv2_b, bn2_g, bn2_b, bn2_m, bn2_v,
                                w2f, cst);
  int zn = NB * KK * CC + NB * KK;  // vl and asum are contiguous
  k_zero<<<(zn + 255) / 256, 256, 0, stream>>>(vl, zn);
  k_norm<<<NB * TT, CC, 0, stream>>>(x, xnh);
  k_assign<<<dim3(TT / 16, NB), 32, 0, stream>>>(xnh, w2f, cst, length, ah, asum);
  k_vlad<<<dim3(TT / 512, NB), 128, 0, stream>>>(ah, xnh, vl);
  k_final<<<NB, CC, 0, stream>>>(vl, asum, centroids, out);
}
